// DeepSeekBlock_3367254360091
// MI455X (gfx1250) — matrix operand order repaired and checked by reading, not compiled, not run on hardware
//
#include <hip/hip_runtime.h>
#include <hip/hip_bf16.h>

typedef _Float16 h16;
typedef unsigned int u32;
typedef __attribute__((ext_vector_type(16))) _Float16 v16h;
typedef __attribute__((ext_vector_type(8)))  float    v8f;
typedef __attribute__((ext_vector_type(4)))  u32      v4u;
typedef __attribute__((ext_vector_type(8)))  int      v8i;
typedef __attribute__((ext_vector_type(4)))  int      v4i;

#define WMMA16(A_, B_, C_) \
  __builtin_amdgcn_wmma_f32_16x16x32_f16(false, (A_), false, (B_), (short)0, (C_), false, false)

// ---- Tensor Data Mover (gfx1250) -------------------------------------------
#ifndef USE_TDM
#if defined(__has_builtin)
#if __has_builtin(__builtin_amdgcn_tensor_load_to_lds)
#define USE_TDM 1
#else
#define USE_TDM 0
#endif
#else
#define USE_TDM 0
#endif
#endif

#if USE_TDM
// 2-D tile load: rows x row_elems (f16), global row stride row_stride_elems,
// LDS dest padded by 1 DWORD every 16 DWORDs (-> 34-half row stride = LDA_S).
__device__ __forceinline__ void tdm_load_tile_2d(const h16* gsrc, u32 lds_byte_off,
                                                 u32 rows, u32 row_elems,
                                                 u32 row_stride_elems) {
  unsigned long long ga = (unsigned long long)(size_t)gsrc;
  v4u g0 = { 1u,                                   // count=1, user descriptor
             lds_byte_off,                         // D#.lds_addr (bytes)
             (u32)ga,                              // global_addr[31:0]
             (u32)(ga >> 32) | (2u << 30) };       // global_addr[56:32] | type=2
  u32 w0 = (1u << 16)                              // data_size = 2 bytes
         | (1u << 20)                              // pad_enable
         | (3u << 22)                              // pad_interval: 16 DWORDs
         | (0u << 25);                             // pad_amount: 1 DWORD
  v8i g1 = { (int)w0,
             (int)((row_elems & 0xFFFFu) << 16),                        // tensor_dim0[15:0]
             (int)((row_elems >> 16) | ((rows & 0xFFFFu) << 16)),      // dim0[31:16]|dim1[15:0]
             (int)((rows >> 16) | ((row_elems & 0xFFFFu) << 16)),      // dim1[31:16]|tile_dim0
             (int)(rows & 0xFFFFu),                                    // tile_dim1 (tile_dim2=0)
             (int)row_stride_elems,                                    // tensor_dim0_stride lo
             0, 0 };
  v4i gz4 = { 0, 0, 0, 0 };
  v8i gz8 = { 0, 0, 0, 0, 0, 0, 0, 0 };
  // clang-23 / therock arity: (g0, g1, g2, g3, extra, cpol)
  __builtin_amdgcn_tensor_load_to_lds(g0, g1, gz4, gz4, gz8, 0);
}
#endif

union F16Frag { v16h v; u32 u[8]; };
union U4H8   { v4u u4; h16 h[8]; u32 u[4]; };

constexpr int T_B = 4, T_S = 2048, T_D = 576, T_H = 9, T_HD = 64, T_DC = 64, T_FF = 1536, T_E = 8;
constexpr int T_M = T_B * T_S;
constexpr int BM = 128, BN = 64, BK = 32;
constexpr int LDA_S = BK + 2;   // 34 halfs: matches TDM pad (16 DW + 1 DW pad)
constexpr int LDB_S = BK + 2;

// ---- fragment loaders -------------------------------------------------------
// A frag (16x32 f16): lane = 16*lh + m ; a[i] = A[m][16*lh + i]
__device__ __forceinline__ void lds_a_frag(F16Frag& f, const h16* row, int lh) {
  const u32* r = (const u32*)row;
#pragma unroll
  for (int t = 0; t < 4; ++t) { f.u[t] = r[8 * lh + t]; f.u[4 + t] = r[8 * lh + 4 + t]; }
}
// B frag (32x16 f16): lane = 16*lh + n ; b[i] = B[16*lh + i][n]; LDS holds B transposed [n][k]
__device__ __forceinline__ void lds_b_frag(F16Frag& f, const h16* row, int lh) {
  const u32* r = (const u32*)row;
#pragma unroll
  for (int t = 0; t < 8; ++t) f.u[t] = r[8 * lh + t];
}
// global: two 16B chunks -> one fragment
__device__ __forceinline__ void g_pair16(F16Frag& f, const h16* lo, const h16* hi) {
  U4H8 a, b; a.u4 = *(const v4u*)lo; b.u4 = *(const v4u*)hi;
#pragma unroll
  for (int j = 0; j < 4; ++j) { f.u[j] = a.u[j]; f.u[4 + j] = b.u[j]; }
}

// ---- f32 -> f16 convert -----------------------------------------------------
__global__ void cvt_f32_f16_kernel(const float* __restrict__ s, h16* __restrict__ d, size_t n) {
  size_t i = (size_t)blockIdx.x * blockDim.x + threadIdx.x;
  size_t st = (size_t)gridDim.x * blockDim.x;
  for (; i < n; i += st) d[i] = (h16)s[i];
}

// ---- RMSNorm ----------------------------------------------------------------
__global__ __launch_bounds__(256) void rmsnorm_kernel(const float* __restrict__ X, h16* __restrict__ O) {
  int row = blockIdx.x;
  const float* xr = X + (size_t)row * T_D;
  float ss = 0.f;
  for (int c = threadIdx.x; c < T_D; c += 256) { float v = xr[c]; ss += v * v; }
#pragma unroll
  for (int m = 16; m >= 1; m >>= 1) ss += __shfl_xor(ss, m, 32);
  __shared__ float red[8];
  __shared__ float invs;
  int wid = threadIdx.x >> 5, lane = threadIdx.x & 31;
  if (lane == 0) red[wid] = ss;
  __syncthreads();
  if (threadIdx.x == 0) {
    float t = 0.f;
#pragma unroll
    for (int j = 0; j < 8; ++j) t += red[j];
    invs = rsqrtf(t / (float)T_D + 1e-6f);
  }
  __syncthreads();
  float iv = invs;
  for (int c = threadIdx.x; c < T_D; c += 256) O[(size_t)row * T_D + c] = (h16)(xr[c] * iv);
}

// ---- GEMM: C(f32, MxN) = A(f16, MxK) @ B(f16, KxN) [+ add0] -----------------
__global__ __launch_bounds__(256)
void gemm_f16_wmma_kernel(const h16* __restrict__ A, const h16* __restrict__ Bw,
                          const float* add0, float* C, int N, int K) {
  __shared__ __align__(16) h16 As[BM * LDA_S];
  __shared__ __align__(16) h16 Bt[BN * LDB_S];
  const int tid = threadIdx.x;
  const int wave = tid >> 5, lane = tid & 31;
  const int wm = wave >> 1, wn = wave & 1;
  const int lh = lane >> 4, ln = lane & 15;
  const int m0 = blockIdx.y * BM, n0 = blockIdx.x * BN;
  const int bk = tid >> 3, bc0 = (tid & 7) * 8;
#if USE_TDM
  const u32 asBase = (u32)(size_t)(&As[0]);
#else
  const int arow = tid >> 1, aseg = tid & 1;
#endif

  v8f acc[2][2] = {};

  for (int k0 = 0; k0 < K; k0 += BK) {
#if USE_TDM
    if (wave == 0)   // one TDM op stages the whole 128x32 A tile (padded rows)
      tdm_load_tile_2d(A + (size_t)m0 * K + k0, asBase, BM, BK, (u32)K);
#else
    { // stage A tile 128x32 (16 halfs / thread)
      U4H8 p0, p1;
      const v4u* g = (const v4u*)(A + (size_t)(m0 + arow) * K + k0 + aseg * 16);
      p0.u4 = g[0]; p1.u4 = g[1];
      u32* d = (u32*)&As[arow * LDA_S + aseg * 16];
#pragma unroll
      for (int j = 0; j < 4; ++j) { d[j] = p0.u[j]; d[4 + j] = p1.u[j]; }
    }
#endif
    { // stage B tile 32x64, transposed into Bt[n][k]
      U4H8 pb;
      pb.u4 = *(const v4u*)(Bw + (size_t)(k0 + bk) * N + n0 + bc0);
#pragma unroll
      for (int j = 0; j < 8; ++j) Bt[(bc0 + j) * LDB_S + bk] = pb.h[j];
    }
    if (k0 + BK < K)   // speculative prefetch of next B tile -> global_prefetch_b8
      __builtin_prefetch(Bw + (size_t)(k0 + BK + bk) * N + n0 + bc0, 0, 1);
#if USE_TDM
    if (wave == 0) __builtin_amdgcn_s_wait_tensorcnt(0);
#endif
    __syncthreads();
    F16Frag af[2], bf[2];
#pragma unroll
    for (int mi = 0; mi < 2; ++mi) lds_a_frag(af[mi], &As[(wm * 32 + mi * 16 + ln) * LDA_S], lh);
#pragma unroll
    for (int ni = 0; ni < 2; ++ni) lds_b_frag(bf[ni], &Bt[(wn * 32 + ni * 16 + ln) * LDB_S], lh);
#pragma unroll
    for (int mi = 0; mi < 2; ++mi)
#pragma unroll
      for (int ni = 0; ni < 2; ++ni) acc[mi][ni] = WMMA16(af[mi].v, bf[ni].v, acc[mi][ni]);
    __syncthreads();
  }

#pragma unroll
  for (int mi = 0; mi < 2; ++mi)
#pragma unroll
    for (int ni = 0; ni < 2; ++ni)
#pragma unroll
      for (int g = 0; g < 8; ++g) {
        int r = m0 + wm * 32 + mi * 16 + g + 8 * lh;   // C/D layout: row = g + 8*(lane>=16)
        int c = n0 + wn * 32 + ni * 16 + ln;
        size_t idx = (size_t)r * N + c;
        float v = acc[mi][ni][g];
        if (add0) v += add0[idx];
        C[idx] = v;
      }
}

// ---- dual gated GEMM: H(f16) = scale[m] * gelu(A@B1) * (A@B3) ---------------
__global__ __launch_bounds__(256)
void gemm_dual_gated_kernel(const h16* __restrict__ A, const h16* __restrict__ B1,
                            const h16* __restrict__ B3, const float* scale, int scaleStride,
                            h16* __restrict__ Hout, int N, int K) {
  __shared__ __align__(16) h16 As[BM * LDA_S];
  __shared__ __align__(16) h16 Bt1[BN * LDB_S];
  __shared__ __align__(16) h16 Bt3[BN * LDB_S];
  const int tid = threadIdx.x;
  const int wave = tid >> 5, lane = tid & 31;
  const int wm = wave >> 1, wn = wave & 1;
  const int lh = lane >> 4, ln = lane & 15;
  const int m0 = blockIdx.y * BM, n0 = blockIdx.x * BN;
  const int bk = tid >> 3, bc0 = (tid & 7) * 8;
#if USE_TDM
  const u32 asBase = (u32)(size_t)(&As[0]);
#else
  const int arow = tid >> 1, aseg = tid & 1;
#endif

  v8f acc1[2][2] = {}, acc3[2][2] = {};

  for (int k0 = 0; k0 < K; k0 += BK) {
#if USE_TDM
    if (wave == 0)
      tdm_load_tile_2d(A + (size_t)m0 * K + k0, asBase, BM, BK, (u32)K);
#else
    {
      U4H8 p0, p1;
      const v4u* g = (const v4u*)(A + (size_t)(m0 + arow) * K + k0 + aseg * 16);
      p0.u4 = g[0]; p1.u4 = g[1];
      u32* d = (u32*)&As[arow * LDA_S + aseg * 16];
#pragma unroll
      for (int j = 0; j < 4; ++j) { d[j] = p0.u[j]; d[4 + j] = p1.u[j]; }
    }
#endif
    {
      U4H8 pa, pc;
      pa.u4 = *(const v4u*)(B1 + (size_t)(k0 + bk) * N + n0 + bc0);
      pc.u4 = *(const v4u*)(B3 + (size_t)(k0 + bk) * N + n0 + bc0);
#pragma unroll
      for (int j = 0; j < 8; ++j) {
        Bt1[(bc0 + j) * LDB_S + bk] = pa.h[j];
        Bt3[(bc0 + j) * LDB_S + bk] = pc.h[j];
      }
    }
    if (k0 + BK < K) {
      __builtin_prefetch(B1 + (size_t)(k0 + BK + bk) * N + n0 + bc0, 0, 1);
      __builtin_prefetch(B3 + (size_t)(k0 + BK + bk) * N + n0 + bc0, 0, 1);
    }
#if USE_TDM
    if (wave == 0) __builtin_amdgcn_s_wait_tensorcnt(0);
#endif
    __syncthreads();
    F16Frag af[2], bf1[2], bf3[2];
#pragma unroll
    for (int mi = 0; mi < 2; ++mi) lds_a_frag(af[mi], &As[(wm * 32 + mi * 16 + ln) * LDA_S], lh);
#pragma unroll
    for (int ni = 0; ni < 2; ++ni) {
      lds_b_frag(bf1[ni], &Bt1[(wn * 32 + ni * 16 + ln) * LDB_S], lh);
      lds_b_frag(bf3[ni], &Bt3[(wn * 32 + ni * 16 + ln) * LDB_S], lh);
    }
#pragma unroll
    for (int mi = 0; mi < 2; ++mi)
#pragma unroll
      for (int ni = 0; ni < 2; ++ni) {
        acc1[mi][ni] = WMMA16(af[mi].v, bf1[ni].v, acc1[mi][ni]);
        acc3[mi][ni] = WMMA16(af[mi].v, bf3[ni].v, acc3[mi][ni]);
      }
    __syncthreads();
  }

#pragma unroll
  for (int mi = 0; mi < 2; ++mi)
#pragma unroll
    for (int ni = 0; ni < 2; ++ni)
#pragma unroll
      for (int g = 0; g < 8; ++g) {
        int r = m0 + wm * 32 + mi * 16 + g + 8 * lh;
        int c = n0 + wn * 32 + ni * 16 + ln;
        float a1 = acc1[mi][ni][g], a3 = acc3[mi][ni][g];
        float ge = 0.5f * a1 * (1.0f + erff(a1 * 0.70710678118654752f));   // exact gelu
        float sc = scale ? scale[(size_t)r * scaleStride] : 1.0f;
        Hout[(size_t)r * N + c] = (h16)(sc * ge * a3);
      }
}

// ---- RoPE + head split ------------------------------------------------------
__global__ __launch_bounds__(64)
void rope_split_kernel(const float* __restrict__ Qf, const float* __restrict__ KVf,
                       h16* __restrict__ Q16, h16* __restrict__ K16, h16* __restrict__ VT16) {
  int idx = blockIdx.x;                 // b*S*H + s*H + h
  int hh = idx % T_H;
  int s  = (idx / T_H) % T_S;
  int b  = idx / (T_H * T_S);
  int d  = threadIdx.x;
  const float* qr = Qf + (size_t)(b * T_S + s) * T_D + hh * T_HD;
  int i = d >> 1;
  float theta = powf(10000.0f, -(float)(2 * i) / 64.0f);
  float fr = (float)s * theta;
  float cs = cosf(fr), sn = sinf(fr);
  float qe = qr[2 * i], qo = qr[2 * i + 1];
  float val = (d & 1) ? (qe * sn + qo * cs) : (qe * cs - qo * sn);
  size_t hidx = ((size_t)(b * T_H + hh) * T_S + s) * T_HD + d;
  Q16[hidx] = (h16)(val * 0.125f);      // fold 1/sqrt(64) exactly
  const float* kvr = KVf + (size_t)(b * T_S + s) * (T_H * 2 * T_DC) + hh * (2 * T_DC);
  K16[hidx] = (h16)kvr[d];
  VT16[((size_t)(b * T_H + hh) * T_HD + d) * T_S + s] = (h16)kvr[T_DC + d];
}

// ---- flash attention: 1 wave per 16 query rows ------------------------------
__global__ __launch_bounds__(256)
void attn_kernel(const h16* __restrict__ Q, const h16* __restrict__ Kd,
                 const h16* __restrict__ Vt, h16* __restrict__ AO) {
  const int b = blockIdx.z, hh = blockIdx.y;
  const int wave = threadIdx.x >> 5, lane = threadIdx.x & 31;
  const int lh = lane >> 4, ln = lane & 15;
  const int q0 = (blockIdx.x * 8 + wave) * 16;
  const h16* qb = Q + (size_t)(b * T_H + hh) * T_S * T_HD;
  const h16* kb = Kd + (size_t)(b * T_H + hh) * T_S * T_HD;
  const h16* vb = Vt + (size_t)(b * T_H + hh) * T_HD * T_S;
  __shared__ __align__(16) h16 Pl[8][16 * 34];
  h16* Pw = Pl[wave];

  F16Frag aq[2];
#pragma unroll
  for (int dc = 0; dc < 2; ++dc) {
    const h16* base = qb + (size_t)(q0 + ln) * T_HD + dc * 32 + 16 * lh;
    g_pair16(aq[dc], base, base + 8);
  }

  float rmax[8], rsum[8];
  v8f O[4] = {};
#pragma unroll
  for (int g = 0; g < 8; ++g) { rmax[g] = -3.0e38f; rsum[g] = 0.f; }

  for (int t0 = 0; t0 < T_S; t0 += 32) {
    v8f sc0 = {}, sc1 = {};
#pragma unroll
    for (int dc = 0; dc < 2; ++dc) {
      F16Frag kb0, kb1;
      const h16* k0p = kb + (size_t)(t0 + ln) * T_HD + dc * 32 + 16 * lh;
      const h16* k1p = kb + (size_t)(t0 + 16 + ln) * T_HD + dc * 32 + 16 * lh;
      g_pair16(kb0, k0p, k0p + 8);
      g_pair16(kb1, k1p, k1p + 8);
      sc0 = WMMA16(aq[dc].v, kb0.v, sc0);
      sc1 = WMMA16(aq[dc].v, kb1.v, sc1);
    }
    float pv0[8], pv1[8], scl[8];
#pragma unroll
    for (int g = 0; g < 8; ++g) {
      float m = fmaxf(sc0[g], sc1[g]);
      m = fmaxf(m, __shfl_xor(m, 1, 16));
      m = fmaxf(m, __shfl_xor(m, 2, 16));
      m = fmaxf(m, __shfl_xor(m, 4, 16));
      m = fmaxf(m, __shfl_xor(m, 8, 16));
      float nm = fmaxf(rmax[g], m);
      float s = __expf(rmax[g] - nm);
      rmax[g] = nm;
      float p0 = __expf(sc0[g] - nm);
      float p1 = __expf(sc1[g] - nm);
      float rs = p0 + p1;
      rs += __shfl_xor(rs, 1, 16);
      rs += __shfl_xor(rs, 2, 16);
      rs += __shfl_xor(rs, 4, 16);
      rs += __shfl_xor(rs, 8, 16);
      rsum[g] = rsum[g] * s + rs;
      scl[g] = s; pv0[g] = p0; pv1[g] = p1;
    }
#pragma unroll
    for (int ni = 0; ni < 4; ++ni)
#pragma unroll
      for (int g = 0; g < 8; ++g) O[ni][g] *= scl[g];
    // transpose P (score layout -> A-frag layout) through per-wave LDS
#pragma unroll
    for (int g = 0; g < 8; ++g) {
      Pw[(g + 8 * lh) * 34 + ln] = (h16)pv0[g];
      Pw[(g + 8 * lh) * 34 + 16 + ln] = (h16)pv1[g];
    }
    __builtin_amdgcn_wave_barrier();
    F16Frag ap;
    lds_a_frag(ap, &Pw[ln * 34], lh);
#pragma unroll
    for (int ni = 0; ni < 4; ++ni) {
      F16Frag bv;
      const h16* vr = vb + (size_t)(ni * 16 + ln) * T_S + t0 + 16 * lh;
      g_pair16(bv, vr, vr + 8);
      O[ni] = WMMA16(ap.v, bv.v, O[ni]);
    }
  }
#pragma unroll
  for (int ni = 0; ni < 4; ++ni)
#pragma unroll
    for (int g = 0; g < 8; ++g) {
      int srow = q0 + g + 8 * lh;
      float o = O[ni][g] / rsum[g];
      AO[(size_t)(b * T_S + srow) * T_D + hh * T_HD + ni * 16 + ln] = (h16)o;
    }
}

// ---- gate + top-2 -----------------------------------------------------------
__global__ __launch_bounds__(256)
void gate_topk_kernel(const h16* __restrict__ XN, const float* __restrict__ GW,
                      float* __restrict__ comb) {
  int t = threadIdx.x;
  int token = blockIdx.x * 32 + (t >> 3);
  int e = t & 7;
  const h16* xr = XN + (size_t)token * T_D;
  float acc = 0.f;
  for (int d = 0; d < T_D; ++d) acc += (float)xr[d] * GW[d * T_E + e];
  __shared__ float sl[256];
  sl[t] = acc;
  __syncthreads();
  int base = t & ~7;
  float l[8]; float mx = -3e38f;
#pragma unroll
  for (int j = 0; j < 8; ++j) { l[j] = sl[base + j]; mx = fmaxf(mx, l[j]); }
  float se = 0.f;
#pragma unroll
  for (int j = 0; j < 8; ++j) { l[j] = __expf(l[j] - mx); se += l[j]; }
#pragma unroll
  for (int j = 0; j < 8; ++j) l[j] /= se;
  int i0 = 0;
#pragma unroll
  for (int j = 1; j < 8; ++j) if (l[j] > l[i0]) i0 = j;
  int i1 = (i0 == 0) ? 1 : 0;
#pragma unroll
  for (int j = 0; j < 8; ++j) if (j != i0 && l[j] > l[i1]) i1 = j;
  float ts = l[i0] + l[i1];
  float val = (e == i0) ? l[i0] / ts : (e == i1) ? l[i1] / ts : 0.f;
  comb[(size_t)token * T_E + e] = val;
}

// ---- orchestration ----------------------------------------------------------
extern "C" void kernel_launch(void* const* d_in, const int* in_sizes, int n_in,
                              void* d_out, int out_size, void* d_ws, size_t ws_size,
                              hipStream_t stream) {
  (void)in_sizes; (void)n_in; (void)out_size; (void)ws_size;
  const float* x      = (const float*)d_in[0];
  const float* q_w    = (const float*)d_in[1];
  const float* kv_w   = (const float*)d_in[2];
  const float* o_w    = (const float*)d_in[3];
  const float* gate_w = (const float*)d_in[4];
  const float* w1     = (const float*)d_in[5];
  const float* w2     = (const float*)d_in[6];
  const float* w3     = (const float*)d_in[7];
  const float* sw1    = (const float*)d_in[8];
  const float* sw2    = (const float*)d_in[9];
  const float* sw3    = (const float*)d_in[10];
  float* out = (float*)d_out;

  char* wsp = (char*)d_ws;
  size_t off = 0;
  auto alloc = [&](size_t bytes) -> char* {
    char* p = wsp + off;
    off = (off + bytes + 255) & ~(size_t)255;
    return p;
  };
  h16* wq16  = (h16*)alloc((size_t)T_D * T_D * 2);
  h16* wkv16 = (h16*)alloc((size_t)T_D * T_H * 2 * T_DC * 2);
  h16* wo16  = (h16*)alloc((size_t)T_D * T_D * 2);
  h16* w116  = (h16*)alloc((size_t)T_E * T_D * T_FF * 2);
  h16* w216  = (h16*)alloc((size_t)T_E * T_FF * T_D * 2);
  h16* w316  = (h16*)alloc((size_t)T_E * T_D * T_FF * 2);
  h16* sw116 = (h16*)alloc((size_t)T_D * T_FF * 2);
  h16* sw216 = (h16*)alloc((size_t)T_FF * T_D * 2);
  h16* sw316 = (h16*)alloc((size_t)T_D * T_FF * 2);
  h16* xn1   = (h16*)alloc((size_t)T_M * T_D * 2);
  float* qf  = (float*)alloc((size_t)T_M * T_D * 4);
  float* kvf = (float*)alloc((size_t)T_M * T_H * 2 * T_DC * 4);
  h16* q16   = (h16*)alloc((size_t)T_M * T_D * 2);
  h16* k16   = (h16*)alloc((size_t)T_M * T_D * 2);
  h16* vt16  = (h16*)alloc((size_t)T_M * T_D * 2);
  h16* ao16  = (h16*)alloc((size_t)T_M * T_D * 2);
  h16* xn2   = (h16*)alloc((size_t)T_M * T_D * 2);
  float* comb = (float*)alloc((size_t)T_M * T_E * 4);
  h16* He    = (h16*)alloc((size_t)T_M * T_FF * 2);

  auto cvt = [&](const float* s, h16* d, size_t n) {
    int blocks = (int)((n + 1023) / 1024);
    if (blocks > 4096) blocks = 4096;
    if (blocks < 1) blocks = 1;
    cvt_f32_f16_kernel<<<blocks, 256, 0, stream>>>(s, d, n);
  };
  cvt(q_w,  wq16,  (size_t)T_D * T_D);
  cvt(kv_w, wkv16, (size_t)T_D * T_H * 2 * T_DC);
  cvt(o_w,  wo16,  (size_t)T_D * T_D);
  cvt(w1,   w116,  (size_t)T_E * T_D * T_FF);
  cvt(w2,   w216,  (size_t)T_E * T_FF * T_D);
  cvt(w3,   w316,  (size_t)T_E * T_D * T_FF);
  cvt(sw1,  sw116, (size_t)T_D * T_FF);
  cvt(sw2,  sw216, (size_t)T_FF * T_D);
  cvt(sw3,  sw316, (size_t)T_D * T_FF);

  rmsnorm_kernel<<<T_M, 256, 0, stream>>>(x, xn1);

  dim3 gD(T_D / BN, T_M / BM);                 // N=576
  dim3 gKV((T_H * 2 * T_DC) / BN, T_M / BM);   // N=1152
  dim3 gFF(T_FF / BN, T_M / BM);               // N=1536
  gemm_f16_wmma_kernel<<<gD, 256, 0, stream>>>(xn1, wq16, nullptr, qf, T_D, T_D);
  gemm_f16_wmma_kernel<<<gKV, 256, 0, stream>>>(xn1, wkv16, nullptr, kvf, T_H * 2 * T_DC, T_D);

  rope_split_kernel<<<T_B * T_S * T_H, 64, 0, stream>>>(qf, kvf, q16, k16, vt16);

  dim3 ga(T_S / 128, T_H, T_B);
  attn_kernel<<<ga, 256, 0, stream>>>(q16, k16, vt16, ao16);

  // x2 = ao @ o_w + x   (into d_out)
  gemm_f16_wmma_kernel<<<gD, 256, 0, stream>>>(ao16, wo16, x, out, T_D, T_D);

  rmsnorm_kernel<<<T_M, 256, 0, stream>>>(out, xn2);
  gate_topk_kernel<<<T_M / 32, 256, 0, stream>>>(xn2, gate_w, comb);

  for (int e = 0; e < T_E; ++e) {
    gemm_dual_gated_kernel<<<gFF, 256, 0, stream>>>(
        xn2, w116 + (size_t)e * T_D * T_FF, w316 + (size_t)e * T_D * T_FF,
        comb + e, T_E, He, T_FF, T_D);
    gemm_f16_wmma_kernel<<<gD, 256, 0, stream>>>(
        He, w216 + (size_t)e * T_FF * T_D, out, out, T_D, T_FF);
  }
  // shared expert
  gemm_dual_gated_kernel<<<gFF, 256, 0, stream>>>(xn2, sw116, sw316, nullptr, 0, He, T_FF, T_D);
  gemm_f16_wmma_kernel<<<gD, 256, 0, stream>>>(He, sw216, out, out, T_D, T_FF);
}
